// MultiheadAttention_80616536146610
// MI455X (gfx1250) — matrix operand order repaired and checked by reading, not compiled, not run on hardware
//
#include <hip/hip_runtime.h>
#include <hip/hip_bf16.h>

#define SLEN 8192
#define NBATCH 4
#define NC 512
#define NH 8
#define ND 64

typedef __attribute__((ext_vector_type(16))) __bf16 v16bf;
typedef __attribute__((ext_vector_type(8)))  __bf16 v8bf;
typedef __attribute__((ext_vector_type(8)))  float  v8f;

union V16U { v16bf v; v8bf h[2]; };

// Native bf16 converts: compiler emits hardware cvt (RNE) instead of manual
// bit-twiddling VALU sequences.
__device__ __forceinline__ __bf16 f2bf(float f) { return (__bf16)f; }
__device__ __forceinline__ float bf2f(__bf16 b) { return (float)b; }

__device__ __forceinline__ v8bf cvt8(float4 a, float4 b) {
  v8bf r;
  r[0] = (__bf16)a.x; r[1] = (__bf16)a.y; r[2] = (__bf16)a.z; r[3] = (__bf16)a.w;
  r[4] = (__bf16)b.x; r[5] = (__bf16)b.y; r[6] = (__bf16)b.z; r[7] = (__bf16)b.w;
  return r;
}

__device__ __forceinline__ v8f wmma_bf16(v16bf a, v16bf b, v8f c) {
  // D = A(16x32 bf16) * B(32x16 bf16) + C(16x16 f32)
  return __builtin_amdgcn_wmma_f32_16x16x32_bf16(false, a, false, b, (short)0, c,
                                                 false, false);
}

__device__ __forceinline__ v8f v8f_zero() {
  v8f z = {0.f, 0.f, 0.f, 0.f, 0.f, 0.f, 0.f, 0.f};
  return z;
}

// ---- gfx1250 async global->LDS staging (ASYNCcnt path), with safe fallback --
#if defined(__has_builtin)
#if __has_builtin(__builtin_amdgcn_global_load_async_to_lds_b128) && \
    __has_builtin(__builtin_amdgcn_s_wait_asynccnt)
#define USE_ASYNC_LDS 1
#endif
#endif
#ifndef USE_ASYNC_LDS
#define USE_ASYNC_LDS 0
#endif

#if USE_ASYNC_LDS
// Builtin signature (from hipcc diagnostic): param 1 is
// 'int __vector(4) __device__ *' (AS1), param 2 the LDS-side counterpart.
typedef int b128_t __attribute__((vector_size(16)));
typedef __attribute__((address_space(1))) b128_t* as1_b128;
typedef __attribute__((address_space(3))) b128_t* as3_b128;
#endif

__device__ __forceinline__ void stage16B(const __bf16* src, __bf16* dst) {
#if USE_ASYNC_LDS
  __builtin_amdgcn_global_load_async_to_lds_b128(
      (as1_b128)(void*)src, (as3_b128)(void*)dst, 0, 0);
#else
  *(v8bf*)dst = *(const v8bf*)src;
#endif
}
__device__ __forceinline__ void stage_wait() {
#if USE_ASYNC_LDS
  __builtin_amdgcn_s_wait_asynccnt(0);
#endif
}

// ---------------------------------------------------------------------------
// Weight pre-convert: W f32 [C,C] -> bf16 (once per launch, removes per-WG
// conversion VALU and halves W L2 traffic).
// ---------------------------------------------------------------------------
__global__ __launch_bounds__(256) void cvt_w_kernel(const float* __restrict__ W,
                                                    __bf16* __restrict__ Wb) {
  const int i = (blockIdx.x * 256 + threadIdx.x) * 4;
  float4 f = *(const float4*)(W + i);
  __bf16* d = Wb + i;
  d[0] = f2bf(f.x); d[1] = f2bf(f.y); d[2] = f2bf(f.z); d[3] = f2bf(f.w);
}

// ---------------------------------------------------------------------------
// Phase 1: Y[b,h,n,d] = act( X[n,b,:] @ W^T + bias )  stored bf16 [B,H,S,D]
// Tile: 128 tokens x 256 output cols, K = 512 in steps of 32. 8 waves.
// A fragments straight from global (no reuse -> no LDS); B (weights, reused by
// all 8 waves) staged in LDS via async copy.
// ---------------------------------------------------------------------------
template <int ACT>
__global__ __launch_bounds__(256) void proj_qkv_kernel(
    const float* __restrict__ X, const __bf16* __restrict__ Wb,
    const float* __restrict__ bias, __bf16* __restrict__ out) {
  __shared__ struct {
    alignas(64) __bf16 Bt[256 * 32];  // cols x k
  } sm;
  const int tid = threadIdx.x;
  const int wave = tid >> 5, lane = tid & 31, lrow = lane & 15, hf = lane >> 4;
  const int n0 = blockIdx.x * 128;
  const int b = blockIdx.y;
  const int jbase = blockIdx.z * 256;

  v8f acc[16];
#pragma unroll
  for (int jt = 0; jt < 16; ++jt) acc[jt] = v8f_zero();

  const float* xrow =
      X + ((size_t)(n0 + wave * 16 + lrow) * NBATCH + b) * NC;

  for (int ks = 0; ks < NC; ks += 32) {
    // stage B tile: col j holds Wb[jbase+j][ks..ks+31] (64B per thread)
    {
      const __bf16* wsrc = Wb + (size_t)(jbase + tid) * NC + ks;
      __bf16* wdst = &sm.Bt[tid * 32];
#pragma unroll
      for (int p = 0; p < 4; ++p) stage16B(wsrc + p * 8, wdst + p * 8);
    }
    // A fragment direct from global: row = lane&15, K halves by lane>>4
    V16U A;
    {
      const float* xp = xrow + ks + hf * 16;
      float4 a0 = *(const float4*)(xp);
      float4 a1 = *(const float4*)(xp + 4);
      float4 a2 = *(const float4*)(xp + 8);
      float4 a3 = *(const float4*)(xp + 12);
      A.h[0] = cvt8(a0, a1);
      A.h[1] = cvt8(a2, a3);
    }
    stage_wait();
    __syncthreads();
#pragma unroll
    for (int jt = 0; jt < 16; ++jt) {
      v16bf Bv = *(const v16bf*)&sm.Bt[(jt * 16 + lrow) * 32 + hf * 16];
      acc[jt] = wmma_bf16(A.v, Bv, acc[jt]);
    }
    __syncthreads();
  }
  // epilogue: bias (+ ELU+1), write bf16 into [B,H,S,D]
#pragma unroll
  for (int jt = 0; jt < 16; ++jt) {
    const int j = jbase + jt * 16 + lrow;
    const float bj = bias[j];
    const int h = j >> 6, d = j & 63;
    __bf16* op = out + ((size_t)b * NH + h) * SLEN * ND + d;
#pragma unroll
    for (int i = 0; i < 8; ++i) {
      const int n = n0 + wave * 16 + hf * 8 + i;
      float val = acc[jt][i] + bj;
      if (ACT) val = (val > 0.f) ? (val + 1.f) : __expf(val);
      op[(size_t)n * ND] = f2bf(val);
    }
  }
}

// ---------------------------------------------------------------------------
// Phase 2: kv[bh] = sum_n k[n]^T v[n]  (64x64, f32 atomic split-K),
//          ksum[bh][d] = sum_n k[n][d]
// ---------------------------------------------------------------------------
__global__ __launch_bounds__(256) void kv_kernel(const __bf16* __restrict__ K,
                                                 const __bf16* __restrict__ V,
                                                 float* __restrict__ kv,
                                                 float* __restrict__ ksum) {
  __shared__ struct {
    alignas(64) __bf16 kT[64 * 32];  // d x n-chunk
    alignas(64) __bf16 vT[64 * 32];  // f x n-chunk
    float red[256];
  } sm;
  const int tid = threadIdx.x;
  const int wave = tid >> 5, lane = tid & 31, lrow = lane & 15, hf = lane >> 4;
  const int bh = blockIdx.y;
  const int n0 = blockIdx.x * (SLEN / 8);
  const int sd = tid & 63, sg = tid >> 6;
  const __bf16* kb = K + (size_t)bh * SLEN * ND;
  const __bf16* vb = V + (size_t)bh * SLEN * ND;

  const int dr = wave & 3;         // d-row tile (16 rows)
  const int fc = (wave >> 2) * 2;  // two f-col tiles
  v8f acc0 = v8f_zero(), acc1 = v8f_zero();
  float ksacc = 0.f;

  for (int ch = 0; ch < SLEN / 8; ch += 32) {
#pragma unroll
    for (int i = 0; i < 8; ++i) {  // stage transposed chunks (d fastest)
      const int n = n0 + ch + sg * 8 + i;
      __bf16 kk = kb[(size_t)n * ND + sd];
      sm.kT[sd * 32 + sg * 8 + i] = kk;
      ksacc += bf2f(kk);
      sm.vT[sd * 32 + sg * 8 + i] = vb[(size_t)n * ND + sd];
    }
    __syncthreads();
    V16U A;  // A = k^T : rows are d, K-dim is n
    A.h[0] = *(const v8bf*)&sm.kT[(dr * 16 + lrow) * 32 + hf * 16];
    A.h[1] = *(const v8bf*)&sm.kT[(dr * 16 + lrow) * 32 + hf * 16 + 8];
    v16bf B0 = *(const v16bf*)&sm.vT[(fc * 16 + lrow) * 32 + hf * 16];
    v16bf B1 = *(const v16bf*)&sm.vT[((fc + 1) * 16 + lrow) * 32 + hf * 16];
    acc0 = wmma_bf16(A.v, B0, acc0);
    acc1 = wmma_bf16(A.v, B1, acc1);
    __syncthreads();
  }
  float* kvp = kv + (size_t)bh * ND * ND;
#pragma unroll
  for (int i = 0; i < 8; ++i) {
    const int d = dr * 16 + hf * 8 + i;
    atomicAdd(&kvp[(size_t)d * ND + fc * 16 + lrow], acc0[i]);
    atomicAdd(&kvp[(size_t)d * ND + (fc + 1) * 16 + lrow], acc1[i]);
  }
  sm.red[tid] = ksacc;
  __syncthreads();
  if (sg == 0) {
    float t = sm.red[sd] + sm.red[sd + 64] + sm.red[sd + 128] + sm.red[sd + 192];
    atomicAdd(&ksum[bh * ND + sd], t);
  }
}

// ---------------------------------------------------------------------------
// Phase 3: y[n] = (q[n] @ kv) / (q[n].ksum + eps), stored bf16 [B,H,S,D]
// ---------------------------------------------------------------------------
__global__ __launch_bounds__(256) void attn_kernel(const __bf16* __restrict__ Q,
                                                   const float* __restrict__ kv,
                                                   const float* __restrict__ ksum,
                                                   __bf16* __restrict__ Y) {
  __shared__ struct {
    alignas(64) __bf16 kvT[64 * 64];  // f x d
    float ksl[64];
    float dinv[128];
  } sm;
  const int tid = threadIdx.x;
  const int wave = tid >> 5, lane = tid & 31, lrow = lane & 15, hf = lane >> 4;
  const int bh = blockIdx.y;
  const int n0 = blockIdx.x * 128;
  const float* kvp = kv + (size_t)bh * ND * ND;
  {
    const int f = tid >> 2, dg = tid & 3;
#pragma unroll
    for (int i = 0; i < 16; ++i) {
      const int d = dg * 16 + i;
      sm.kvT[f * 64 + d] = f2bf(kvp[(size_t)d * ND + f]);
    }
  }
  if (tid < 64) sm.ksl[tid] = ksum[bh * ND + tid];
  __syncthreads();
  const __bf16* qb = Q + (size_t)bh * SLEN * ND;
  if (tid < 128) {
    const __bf16* qp = qb + (size_t)(n0 + tid) * ND;
    float dot = 0.f;
#pragma unroll
    for (int d = 0; d < 64; ++d) dot += bf2f(qp[d]) * sm.ksl[d];
    sm.dinv[tid] = 1.f / (dot + 1e-6f);
  }
  __syncthreads();
  v8f acc[4];
#pragma unroll
  for (int ft = 0; ft < 4; ++ft) acc[ft] = v8f_zero();
#pragma unroll
  for (int kd = 0; kd < 64; kd += 32) {
    const __bf16* qp = qb + (size_t)(n0 + wave * 16 + lrow) * ND + kd + hf * 16;
    V16U A;
    A.h[0] = *(const v8bf*)qp;
    A.h[1] = *(const v8bf*)(qp + 8);
#pragma unroll
    for (int ft = 0; ft < 4; ++ft) {
      v16bf Bv = *(const v16bf*)&sm.kvT[(ft * 16 + lrow) * 64 + kd + hf * 16];
      acc[ft] = wmma_bf16(A.v, Bv, acc[ft]);
    }
  }
#pragma unroll
  for (int ft = 0; ft < 4; ++ft) {
#pragma unroll
    for (int i = 0; i < 8; ++i) {
      const int rl = wave * 16 + hf * 8 + i;
      float val = acc[ft][i] * sm.dinv[rl];
      Y[((size_t)bh * SLEN + (n0 + rl)) * ND + ft * 16 + lrow] = f2bf(val);
    }
  }
}

// ---------------------------------------------------------------------------
// Phase 4: out[n,b,:] = y_merged[n] @ Wp^T + bp   (f32 output [S,B,C])
// A fragments direct from bf16 y (no conversion, no LDS); B staged async.
// ---------------------------------------------------------------------------
__global__ __launch_bounds__(256) void proj_out_kernel(
    const __bf16* __restrict__ Y, const __bf16* __restrict__ Wb,
    const float* __restrict__ bias, float* __restrict__ out) {
  __shared__ struct {
    alignas(64) __bf16 Bt[256 * 32];
  } sm;
  const int tid = threadIdx.x;
  const int wave = tid >> 5, lane = tid & 31, lrow = lane & 15, hf = lane >> 4;
  const int n0 = blockIdx.x * 128;
  const int b = blockIdx.y;
  const int jbase = blockIdx.z * 256;

  v8f acc[16];
#pragma unroll
  for (int jt = 0; jt < 16; ++jt) acc[jt] = v8f_zero();

  const int nrow = n0 + wave * 16 + lrow;

  for (int ks = 0; ks < NC; ks += 32) {
    {
      const __bf16* wsrc = Wb + (size_t)(jbase + tid) * NC + ks;
      __bf16* wdst = &sm.Bt[tid * 32];
#pragma unroll
      for (int p = 0; p < 4; ++p) stage16B(wsrc + p * 8, wdst + p * 8);
    }
    // A from y with head merge: c = h*64 + d ; a 32-wide K chunk stays in one h
    V16U A;
    {
      const int h = ks >> 6, d0 = (ks & 63) + hf * 16;
      const __bf16* yp =
          Y + (((size_t)b * NH + h) * SLEN + nrow) * ND + d0;
      A.h[0] = *(const v8bf*)yp;
      A.h[1] = *(const v8bf*)(yp + 8);
    }
    stage_wait();
    __syncthreads();
#pragma unroll
    for (int jt = 0; jt < 16; ++jt) {
      v16bf Bv = *(const v16bf*)&sm.Bt[(jt * 16 + lrow) * 32 + hf * 16];
      acc[jt] = wmma_bf16(A.v, Bv, acc[jt]);
    }
    __syncthreads();
  }
#pragma unroll
  for (int jt = 0; jt < 16; ++jt) {
    const int j = jbase + jt * 16 + lrow;
    const float bj = bias[j];
#pragma unroll
    for (int i = 0; i < 8; ++i) {
      const int n = n0 + wave * 16 + hf * 8 + i;
      out[((size_t)n * NBATCH + b) * NC + j] = acc[jt][i] + bj;
    }
  }
}

__global__ void zero_kernel(float* __restrict__ p, int n) {
  int i = blockIdx.x * blockDim.x + threadIdx.x;
  if (i < n) p[i] = 0.f;
}

extern "C" void kernel_launch(void* const* d_in, const int* in_sizes, int n_in,
                              void* d_out, int out_size, void* d_ws,
                              size_t ws_size, hipStream_t stream) {
  (void)in_sizes; (void)n_in; (void)out_size; (void)ws_size;
  const float* query = (const float*)d_in[0];
  const float* key = (const float*)d_in[1];
  const float* value = (const float*)d_in[2];
  const float* Wq = (const float*)d_in[3];
  const float* bq = (const float*)d_in[4];
  const float* Wk = (const float*)d_in[5];
  const float* bk = (const float*)d_in[6];
  const float* Wv = (const float*)d_in[7];
  const float* bv = (const float*)d_in[8];
  const float* Wp = (const float*)d_in[9];
  const float* bp = (const float*)d_in[10];
  float* out = (float*)d_out;

  const size_t TENS = (size_t)NBATCH * NH * SLEN * ND;  // 16777216 elems
  const size_t WEL = (size_t)NC * NC;                   // 262144 elems
  __bf16* qws = (__bf16*)d_ws;
  __bf16* kws = qws + TENS;
  __bf16* vws = kws + TENS;
  __bf16* yws = vws + TENS;
  float* kvw = (float*)(yws + TENS);                 // [B*H,64,64] f32
  float* ksw = kvw + (size_t)NBATCH * NH * ND * ND;  // [B*H,64] f32
  __bf16* wqb = (__bf16*)(ksw + (size_t)NBATCH * NH * ND);
  __bf16* wkb = wqb + WEL;
  __bf16* wvb = wkb + WEL;
  __bf16* wpb = wvb + WEL;

  const dim3 cg((unsigned)(WEL / (256 * 4)));  // 256 blocks
  cvt_w_kernel<<<cg, 256, 0, stream>>>(Wq, wqb);
  cvt_w_kernel<<<cg, 256, 0, stream>>>(Wk, wkb);
  cvt_w_kernel<<<cg, 256, 0, stream>>>(Wv, wvb);
  cvt_w_kernel<<<cg, 256, 0, stream>>>(Wp, wpb);

  const int nzero = NBATCH * NH * ND * ND + NBATCH * NH * ND;  // 133120
  zero_kernel<<<dim3((nzero + 255) / 256), 256, 0, stream>>>(kvw, nzero);

  dim3 pg(SLEN / 128, NBATCH, 2);
  proj_qkv_kernel<1><<<pg, 256, 0, stream>>>(query, wqb, bq, qws);
  proj_qkv_kernel<1><<<pg, 256, 0, stream>>>(key, wkb, bk, kws);
  proj_qkv_kernel<0><<<pg, 256, 0, stream>>>(value, wvb, bv, vws);
  kv_kernel<<<dim3(8, NBATCH * NH), 256, 0, stream>>>(kws, vws, kvw, ksw);
  attn_kernel<<<dim3(SLEN / 128, NBATCH * NH), 256, 0, stream>>>(qws, kvw, ksw,
                                                                 yws);
  proj_out_kernel<<<pg, 256, 0, stream>>>(yws, wpb, bp, out);
}
